// NodeBlock_38345468018711
// MI455X (gfx1250) — matrix operand order repaired and checked by reading, not compiled, not run on hardware
//
#include <hip/hip_runtime.h>
#include <hip/hip_fp16.h>

#define NUM_NODES 100000
#define NUM_EDGES 1600000
#define NODE_DIM  128
#define HIDDEN    128
#define IN_DIM    256   // NODE_DIM + EDGE_DIM
#define LDS_STRIDE 136  // 128 + 8 halves padding (keeps 16B alignment, dodges bank conflicts)

typedef __attribute__((ext_vector_type(16))) _Float16 v16h;
typedef __attribute__((ext_vector_type(8)))  float    v8f;

// ---------------------------------------------------------------------------
// Kernel 1: zero the edge aggregation buffer (must re-zero every call).
// ---------------------------------------------------------------------------
__global__ void zero_aggr_kernel(float4* p, int n4) {
    int i = blockIdx.x * blockDim.x + threadIdx.x;
    if (i < n4) p[i] = make_float4(0.f, 0.f, 0.f, 0.f);
}

// ---------------------------------------------------------------------------
// Kernel 2: convert weights to f16, stored TRANSPOSED: Wh[j][k] = W[k][j].
// This makes every WMMA B-fragment (16 consecutive K values for one output
// column) a single contiguous 32-byte load.
// ---------------------------------------------------------------------------
__global__ void prep_weights_kernel(const float* __restrict__ W0,
                                    const float* __restrict__ W1,
                                    const float* __restrict__ W2,
                                    _Float16* __restrict__ W0h,
                                    _Float16* __restrict__ W1h,
                                    _Float16* __restrict__ W2h) {
    int idx = blockIdx.x * blockDim.x + threadIdx.x;
    const int n0 = IN_DIM * HIDDEN;   // 32768
    const int n1 = HIDDEN * HIDDEN;   // 16384
    if (idx < n0) {
        int k = idx / HIDDEN, j = idx % HIDDEN;
        W0h[j * IN_DIM + k] = (_Float16)W0[idx];
    } else if (idx < n0 + n1) {
        int t = idx - n0; int k = t / HIDDEN, j = t % HIDDEN;
        W1h[j * HIDDEN + k] = (_Float16)W1[t];
    } else if (idx < n0 + 2 * n1) {
        int t = idx - n0 - n1; int k = t / HIDDEN, j = t % HIDDEN;
        W2h[j * HIDDEN + k] = (_Float16)W2[t];
    }
}

// ---------------------------------------------------------------------------
// Kernel 3: scatter-add edge features onto target nodes.
// One wave handles one edge (128 contiguous floats); each lane does a float4
// load + 4 global f32 atomics. Aggregation buffer (51.2MB) lives in L2.
// ---------------------------------------------------------------------------
__global__ void scatter_edges_kernel(const float* __restrict__ edge_attr,
                                     const int*   __restrict__ col,
                                     float*       __restrict__ aggr) {
    long long tid = (long long)blockIdx.x * blockDim.x + threadIdx.x;
    int e = (int)(tid >> 5);
    if (e >= NUM_EDGES) return;
    int c = ((int)tid & 31) * 4;
    int node = col[e];
    const float4 v = *(const float4*)(edge_attr + (size_t)e * 128 + c);
    float* dst = aggr + (size_t)node * 128 + c;
    atomicAdd(dst + 0, v.x);
    atomicAdd(dst + 1, v.y);
    atomicAdd(dst + 2, v.z);
    atomicAdd(dst + 3, v.w);
}

// ---------------------------------------------------------------------------
// Fused MLP + LayerNorm. One wave32 per 16-node tile; 8 waves / block.
// All three GEMMs use v_wmma_f32_16x16x32_f16 with f32 accumulators.
// ---------------------------------------------------------------------------
__device__ __forceinline__ v8f wmma_f16(v16h a, v16h b, v8f c) {
    return __builtin_amdgcn_wmma_f32_16x16x32_f16(
        /*neg_a=*/false, a, /*neg_b=*/false, b,
        /*c_mod=*/(short)0, c, /*reuse_a=*/false, /*reuse_b=*/false);
}

__device__ __forceinline__ void init_acc_bias(v8f acc[8], const float* __restrict__ bias, int m16) {
#pragma unroll
    for (int nt = 0; nt < 8; ++nt) {
        float bj = bias[m16 + 16 * nt];
        v8f c;
#pragma unroll
        for (int i = 0; i < 8; ++i) c[i] = bj;
        acc[nt] = c;
    }
}

// Hidden-layer GEMM with A sourced from this wave's LDS tile (f16, row-major,
// row stride LDS_STRIDE halves). K = HIDDEN = 128 -> 4 k-steps.
__device__ __forceinline__ void layer_from_lds(const _Float16* __restrict__ myh,
                                               const _Float16* __restrict__ Wh,
                                               int m16, int hi, v8f acc[8]) {
#pragma unroll
    for (int step = 0; step < 4; ++step) {
        const int k0 = step * 32;
        const _Float16* ar = myh + m16 * LDS_STRIDE + k0;
        v16h a;
#pragma unroll
        for (int i = 0; i < 8; ++i) {
            a[i]     = ar[hi * 16 + i];       // K = k0 + 16*hi + i
            a[8 + i] = ar[hi * 16 + 8 + i];   // K = k0 + 16*hi + 8 + i
        }
#pragma unroll
        for (int nt = 0; nt < 8; ++nt) {
            const v16h b = *(const v16h*)(Wh + (size_t)(m16 + 16 * nt) * HIDDEN + k0 + hi * 16);
            acc[nt] = wmma_f16(a, b, acc[nt]);
        }
    }
}

// ReLU accumulators -> f16 -> LDS tile (C/D layout: VGPR v <-> M = v + 8*hi).
__device__ __forceinline__ void relu_store_lds(const v8f acc[8], _Float16* __restrict__ myh,
                                               int m16, int hi) {
#pragma unroll
    for (int nt = 0; nt < 8; ++nt) {
#pragma unroll
        for (int v = 0; v < 8; ++v) {
            float r = fmaxf(acc[nt][v], 0.f);
            int m = v + 8 * hi;
            myh[m * LDS_STRIDE + m16 + 16 * nt] = (_Float16)r;
        }
    }
}

__global__ __launch_bounds__(256) void mlp_ln_kernel(
        const float* __restrict__ node_attr, const float* __restrict__ aggr,
        const _Float16* __restrict__ W0h, const _Float16* __restrict__ W1h,
        const _Float16* __restrict__ W2h,
        const float* __restrict__ b0, const float* __restrict__ b1,
        const float* __restrict__ b2,
        const float* __restrict__ ln_g, const float* __restrict__ ln_b,
        float* __restrict__ out) {
    __shared__ _Float16 lds_h[8][16 * LDS_STRIDE];

    const int lane = threadIdx.x & 31;
    const int wave = threadIdx.x >> 5;
    const int m16  = lane & 15;     // node row (A/M index) / output column (B/N index)
    const int hi   = lane >> 4;     // half-wave selector
    const int tile = blockIdx.x * 8 + wave;
    const int nodeBase = tile * 16;

    // Clamped node index for loads; stores are guarded at the end.
    int nodeA = nodeBase + m16;
    if (nodeA >= NUM_NODES) nodeA = NUM_NODES - 1;

    _Float16* myh = &lds_h[wave][0];
    v8f acc[8];

    // ---- Layer 0: x = [node_attr | aggr] (K=256, 8 k-steps), A from global --
    init_acc_bias(acc, b0, m16);
    const float* rowN = node_attr + (size_t)nodeA * 128;
    const float* rowE = aggr      + (size_t)nodeA * 128;
#pragma unroll
    for (int step = 0; step < 8; ++step) {
        const int k0 = step * 32;
        const float* src = (k0 < 128) ? (rowN + k0) : (rowE + (k0 - 128));
        v16h a;
#pragma unroll
        for (int i = 0; i < 8; ++i) {
            a[i]     = (_Float16)src[hi * 16 + i];
            a[8 + i] = (_Float16)src[hi * 16 + 8 + i];
        }
#pragma unroll
        for (int nt = 0; nt < 8; ++nt) {
            const v16h b = *(const v16h*)(W0h + (size_t)(m16 + 16 * nt) * IN_DIM + k0 + hi * 16);
            acc[nt] = wmma_f16(a, b, acc[nt]);
        }
    }
    relu_store_lds(acc, myh, m16, hi);   // h0 -> LDS (f16)

    // ---- Layer 1 (K=128), A from LDS (per-wave region; LDS is in-order) ----
    init_acc_bias(acc, b1, m16);
    layer_from_lds(myh, W1h, m16, hi, acc);
    relu_store_lds(acc, myh, m16, hi);   // h1 -> LDS (reads all precede writes)

    // ---- Layer 2 (K=128): keep result in registers for LayerNorm ----------
    init_acc_bias(acc, b2, m16);
    layer_from_lds(myh, W2h, m16, hi, acc);

    // ---- LayerNorm over the 128 features of each row ----------------------
    // Row m = v + 8*hi lives entirely in one 16-lane half: 8 values per lane
    // (one per n-tile). Reduce across the 16 lanes with 4 xor-shuffles.
    float mu[8], rs[8];
#pragma unroll
    for (int v = 0; v < 8; ++v) {
        float s = 0.f, q = 0.f;
#pragma unroll
        for (int nt = 0; nt < 8; ++nt) { float x = acc[nt][v]; s += x; q += x * x; }
#pragma unroll
        for (int off = 1; off < 16; off <<= 1) {
            s += __shfl_xor(s, off, 32);
            q += __shfl_xor(q, off, 32);
        }
        float mean = s * (1.f / 128.f);
        float var  = q * (1.f / 128.f) - mean * mean;
        mu[v] = mean;
        rs[v] = rsqrtf(var + 1e-5f);
    }

#pragma unroll
    for (int nt = 0; nt < 8; ++nt) {
        const int j = m16 + 16 * nt;
        const float g  = ln_g[j];
        const float bb = ln_b[j];
#pragma unroll
        for (int v = 0; v < 8; ++v) {
            const int node = nodeBase + v + 8 * hi;
            if (node < NUM_NODES)
                out[(size_t)node * 128 + j] = (acc[nt][v] - mu[v]) * rs[v] * g + bb;
        }
    }
}

// ---------------------------------------------------------------------------
extern "C" void kernel_launch(void* const* d_in, const int* in_sizes, int n_in,
                              void* d_out, int out_size, void* d_ws, size_t ws_size,
                              hipStream_t stream) {
    const float* node_attr  = (const float*)d_in[0];
    const float* edge_attr  = (const float*)d_in[1];
    const int*   edge_index = (const int*)d_in[2];   // int32 (jax demotes int64)
    const float* W0 = (const float*)d_in[3];
    const float* b0 = (const float*)d_in[4];
    const float* W1 = (const float*)d_in[5];
    const float* b1 = (const float*)d_in[6];
    const float* W2 = (const float*)d_in[7];
    const float* b2 = (const float*)d_in[8];
    const float* ln_g = (const float*)d_in[9];
    const float* ln_b = (const float*)d_in[10];
    float* out = (float*)d_out;

    // Workspace layout: [aggr f32 51.2MB][W0h 64KB][W1h 32KB][W2h 32KB]
    float* aggr = (float*)d_ws;
    const size_t aggr_bytes = (size_t)NUM_NODES * NODE_DIM * sizeof(float);
    _Float16* W0h = (_Float16*)((char*)d_ws + aggr_bytes);
    _Float16* W1h = W0h + (size_t)IN_DIM * HIDDEN;
    _Float16* W2h = W1h + (size_t)HIDDEN * HIDDEN;

    // 1) zero aggregation buffer (12.8M floats = 3.2M float4)
    {
        const int n4 = NUM_NODES * NODE_DIM / 4;
        zero_aggr_kernel<<<(n4 + 255) / 256, 256, 0, stream>>>((float4*)aggr, n4);
    }
    // 2) f16 transposed weights
    {
        const int total = IN_DIM * HIDDEN + 2 * HIDDEN * HIDDEN;  // 65536
        prep_weights_kernel<<<(total + 255) / 256, 256, 0, stream>>>(W0, W1, W2, W0h, W1h, W2h);
    }
    // 3) scatter-add (one wave per edge)
    {
        const long long work = (long long)NUM_EDGES * 32;
        const int blocks = (int)((work + 255) / 256);   // 200000
        scatter_edges_kernel<<<blocks, 256, 0, stream>>>(edge_attr, edge_index + NUM_EDGES, aggr);
    }
    // 4) fused MLP + LayerNorm (16 nodes per wave, 8 waves per block)
    {
        const int tiles  = (NUM_NODES + 15) / 16;  // 6250
        const int blocks = (tiles + 7) / 8;        // 782
        mlp_ln_kernel<<<blocks, 256, 0, stream>>>(node_attr, aggr, W0h, W1h, W2h,
                                                  b0, b1, b2, ln_g, ln_b, out);
    }
}
